// AttentionDecoder_32384053412363
// MI455X (gfx1250) — matrix operand order repaired and checked by reading, not compiled, not run on hardware
//
#include <hip/hip_runtime.h>
#include <math.h>

// ---- problem constants --------------------------------------------------
constexpr int NB  = 128;    // batch
constexpr int NP  = 196;    // pixels
constexpr int NE  = 2048;   // encoder dim
constexpr int ND  = 512;    // decoder dim
constexpr int NA  = 512;    // attention dim
constexpr int NED = 512;    // embed dim
constexpr int NV  = 10000;  // vocab
constexpr int NL  = 20;     // max caption len
constexpr int NT  = 19;     // decode steps
constexpr int KXH = NED + NE + ND; // 3072 : packed LSTM input [emb, awe, h]
constexpr int G4  = 4 * ND;        // 2048 : gates

typedef __attribute__((ext_vector_type(16))) __bf16 v16bf;
typedef __attribute__((ext_vector_type(8)))  __bf16 v8bf;
typedef __attribute__((ext_vector_type(8)))  float  v8f;

__device__ __forceinline__ float sigf(float x) { return 1.0f / (1.0f + __expf(-x)); }

// ---- epilogue modes -----------------------------------------------------
#define EPI_F32  0
#define EPI_BF16 1
#define EPI_SIG  2

// ---- generic bf16 WMMA GEMM:  C[M,N] = A[M,K] @ Bw[N,K]^T + bias --------
// block = 256 threads (8 waves); block tile = 16 rows x 256 cols
// (2 N-tiles per wave). A staged as 16x64 bf16 tiles, double-buffered in
// LDS (pitch 72 halves -> conflict-free b128 reads). The k-loop is manually
// unrolled by two phases so the A-tile LDS ping-pong and the B-fragment
// register double-buffer are resolved statically (no register rotation
// moves); each phase issues the next tile's global loads before its WMMAs.
// K must be a multiple of 128 (all GEMMs here: 512/2048/3072).
__global__ __launch_bounds__(256)
void k_gemm(const __bf16* __restrict__ A, const __bf16* __restrict__ Bw,
            const float* __restrict__ bias,
            float* __restrict__ Cf, __bf16* __restrict__ Cb,
            int M, int N, int K, long ldc, long coff,
            int epi, const int* __restrict__ declen, int tstep)
{
    constexpr int PITCH = 72;                 // halves per LDS row
    __shared__ __align__(16) __bf16 sA[2][16 * PITCH];
    const int tid  = threadIdx.x;
    const int wid  = tid >> 5;
    const int lane = tid & 31;
    const int hi   = lane >> 4;
    const int m0   = blockIdx.x * 16;
    const int nt0  = (blockIdx.y * 8 + wid) * 2;
    const int n0   = nt0 * 16 + (lane & 15);
    const int n1   = n0 + 16;
    const bool act0 = (nt0 * 16) < N;         // wave-uniform
    const bool act1 = ((nt0 + 1) * 16) < N;

    // staging: thread stages dwords (srow,scol) and (srow+8,scol) of the
    // 16x32-dword (16x64 half) tile
    const int srow = tid >> 5;                // 0..7
    const int scol = tid & 31;                // 0..31 (dword col)
    const uint32_t* as0 = (const uint32_t*)(A + (size_t)(m0 + srow) * K);
    const uint32_t* as1 = (const uint32_t*)(A + (size_t)(m0 + srow + 8) * K);
    uint32_t* sw0 = (uint32_t*)&sA[0][srow * PITCH] + scol;
    uint32_t* sw0b = (uint32_t*)&sA[0][(srow + 8) * PITCH] + scol;
    uint32_t* sw1 = (uint32_t*)&sA[1][srow * PITCH] + scol;
    uint32_t* sw1b = (uint32_t*)&sA[1][(srow + 8) * PITCH] + scol;

    const int nk = K >> 6;                    // even (K multiple of 128)
    uint32_t pv0 = as0[scol];
    uint32_t pv1 = as1[scol];
    *sw0 = pv0; *sw0b = pv1;
    __syncthreads();

    v8f acc0 = {}, acc1 = {};
    const __bf16* bp0 = Bw + (size_t)(act0 ? n0 : 0) * K + (hi ? 16 : 0);
    const __bf16* bp1 = Bw + (size_t)(act1 ? n1 : 0) * K + (hi ? 16 : 0);

    // B fragment sets: X for even k-tiles, Y for odd k-tiles
    v16bf x00, x01, x10, x11, y00, y01, y10, y11;
    if (act0) { x00 = *(const v16bf*)(bp0); x01 = *(const v16bf*)(bp0 + 32); }
    if (act1) { x10 = *(const v16bf*)(bp1); x11 = *(const v16bf*)(bp1 + 32); }

    for (int kt = 0; kt < nk; kt += 2) {
        // ===== even phase: compute tile kt from sA[0] + set X;
        //       prefetch tile kt+1 (A -> sA[1], B -> set Y)
        pv0 = as0[((kt + 1) << 5) + scol];
        pv1 = as1[((kt + 1) << 5) + scol];
        {
            const __bf16* ap = &sA[0][(lane & 15) * PITCH];
            v16bf a0, a1;
            v8bf lo = *(const v8bf*)(ap + (hi ? 16 : 0));
            v8bf hh = *(const v8bf*)(ap + (hi ? 24 : 8));
#pragma unroll
            for (int i = 0; i < 8; ++i) { a0[i] = lo[i]; a0[i + 8] = hh[i]; }
            lo = *(const v8bf*)(ap + 32 + (hi ? 16 : 0));
            hh = *(const v8bf*)(ap + 32 + (hi ? 24 : 8));
#pragma unroll
            for (int i = 0; i < 8; ++i) { a1[i] = lo[i]; a1[i + 8] = hh[i]; }

            if (act0) { y00 = *(const v16bf*)(bp0 + 64); y01 = *(const v16bf*)(bp0 + 96); }
            if (act1) { y10 = *(const v16bf*)(bp1 + 64); y11 = *(const v16bf*)(bp1 + 96); }

            if (act0) {
                acc0 = __builtin_amdgcn_wmma_f32_16x16x32_bf16(false, a0, false, x00, (short)0, acc0, false, false);
                acc0 = __builtin_amdgcn_wmma_f32_16x16x32_bf16(false, a1, false, x01, (short)0, acc0, false, false);
            }
            if (act1) {
                acc1 = __builtin_amdgcn_wmma_f32_16x16x32_bf16(false, a0, false, x10, (short)0, acc1, false, false);
                acc1 = __builtin_amdgcn_wmma_f32_16x16x32_bf16(false, a1, false, x11, (short)0, acc1, false, false);
            }
            *sw1 = pv0; *sw1b = pv1;
            __syncthreads();
        }
        // ===== odd phase: compute tile kt+1 from sA[1] + set Y;
        //       prefetch tile kt+2 (A -> sA[0], B -> set X) if it exists
        {
            const bool more = (kt + 2) < nk;
            if (more) {
                pv0 = as0[((kt + 2) << 5) + scol];
                pv1 = as1[((kt + 2) << 5) + scol];
            }
            const __bf16* ap = &sA[1][(lane & 15) * PITCH];
            v16bf a0, a1;
            v8bf lo = *(const v8bf*)(ap + (hi ? 16 : 0));
            v8bf hh = *(const v8bf*)(ap + (hi ? 24 : 8));
#pragma unroll
            for (int i = 0; i < 8; ++i) { a0[i] = lo[i]; a0[i + 8] = hh[i]; }
            lo = *(const v8bf*)(ap + 32 + (hi ? 16 : 0));
            hh = *(const v8bf*)(ap + 32 + (hi ? 24 : 8));
#pragma unroll
            for (int i = 0; i < 8; ++i) { a1[i] = lo[i]; a1[i + 8] = hh[i]; }

            if (more) {
                if (act0) { x00 = *(const v16bf*)(bp0 + 128); x01 = *(const v16bf*)(bp0 + 160); }
                if (act1) { x10 = *(const v16bf*)(bp1 + 128); x11 = *(const v16bf*)(bp1 + 160); }
            }
            if (act0) {
                acc0 = __builtin_amdgcn_wmma_f32_16x16x32_bf16(false, a0, false, y00, (short)0, acc0, false, false);
                acc0 = __builtin_amdgcn_wmma_f32_16x16x32_bf16(false, a1, false, y01, (short)0, acc0, false, false);
            }
            if (act1) {
                acc1 = __builtin_amdgcn_wmma_f32_16x16x32_bf16(false, a0, false, y10, (short)0, acc1, false, false);
                acc1 = __builtin_amdgcn_wmma_f32_16x16x32_bf16(false, a1, false, y11, (short)0, acc1, false, false);
            }
            if (more) {
                *sw0 = pv0; *sw0b = pv1;
                __syncthreads();
                bp0 += 128; bp1 += 128;
            }
        }
    }

    // epilogue
#pragma unroll
    for (int s = 0; s < 2; ++s) {
        const bool act = s ? act1 : act0;
        if (!act) continue;
        const int n = s ? n1 : n0;
        v8f& acc = s ? acc1 : acc0;
        float bv = bias ? bias[n] : 0.0f;
#pragma unroll
        for (int r = 0; r < 8; ++r) {
            int m = m0 + r + hi * 8;          // C/D layout: M = r + 8*(lane>=16)
            float v = acc[r] + bv;
            if (epi == EPI_BF16) {
                Cb[(size_t)m * N + n] = (__bf16)v;
            } else {
                if (epi == EPI_SIG) v = sigf(v);
                if (declen) v = (tstep < declen[m]) ? v : 0.0f;
                Cf[(size_t)m * ldc + coff + n] = v;
            }
        }
    }
}

// ---- stable descending argsort of caption_len + gathers -----------------
__global__ void k_sort(const int* __restrict__ cap_len, const int* __restrict__ captions,
                       int* order_i, int* declen_i, int* caps_i,
                       float* out_caps, float* out_declen, float* out_order)
{
    __shared__ int slen[NB];
    __shared__ int sord[NB];
    int i = threadIdx.x;
    slen[i] = cap_len[i];
    __syncthreads();
    int li = slen[i], rank = 0;
    for (int j = 0; j < NB; ++j) {
        int lj = slen[j];
        rank += (lj > li) || (lj == li && j < i);
    }
    sord[rank] = i;
    __syncthreads();
    int src = sord[i];
    order_i[i] = src;
    int dl = slen[src] - 1;
    declen_i[i] = dl;
    out_order[i] = (float)src;
    out_declen[i] = (float)dl;
    for (int l = 0; l < NL; ++l) {
        int cv = captions[src * NL + l];
        caps_i[i * NL + l] = cv;
        out_caps[i * NL + l] = (float)cv;
    }
}

// ---- f32 -> bf16 conversion ---------------------------------------------
__global__ void k_cvt(const float* __restrict__ s, __bf16* __restrict__ d, int n)
{
    int i = blockIdx.x * 256 + threadIdx.x;
    if (i < n) d[i] = (__bf16)s[i];
}

// ---- pack [Wih | Whh] -> bf16 [2048, 3072] -------------------------------
__global__ void k_pack_w(const float* __restrict__ Wih, const float* __restrict__ Whh,
                         __bf16* __restrict__ Wpk)
{
    int i = blockIdx.x * 256 + threadIdx.x;   // over 2048*3072
    int nrow = i / KXH, j = i % KXH;
    float v = (j < NED + NE) ? Wih[(size_t)nrow * (NED + NE) + j]
                             : Whh[(size_t)nrow * ND + (j - NED - NE)];
    Wpk[i] = (__bf16)v;
}

__global__ void k_bsum(const float* a, const float* b, float* o, int n)
{
    int i = blockIdx.x * 256 + threadIdx.x;
    if (i < n) o[i] = a[i] + b[i];
}

// ---- gather-sorted encoder -> bf16 (L2-resident working copy) -----------
__global__ void k_enc(const float* __restrict__ enc, const int* __restrict__ order_i,
                      __bf16* __restrict__ enc_b)
{
    size_t i = (size_t)blockIdx.x * 256 + threadIdx.x;
    size_t per = (size_t)NP * NE;
    int b = (int)(i / per);
    size_t r = i % per;
    enc_b[i] = (__bf16)enc[(size_t)order_i[b] * per + r];
}

// ---- mean over P (from f32 original, order-gathered) --------------------
__global__ void k_mean(const float* __restrict__ enc, const int* __restrict__ order_i,
                       __bf16* __restrict__ meanb)
{
    int b = blockIdx.x;
    int e = blockIdx.y * 256 + threadIdx.x;
    const float* base = enc + (size_t)order_i[b] * NP * NE + e;
    float s = 0.0f;
    for (int p = 0; p < NP; ++p) s += base[(size_t)p * NE];
    meanb[b * NE + e] = (__bf16)(s * (1.0f / NP));
}

// ---- e[b,p] = sum_a relu(att1+att2)*fa_w + fa_b  (1 wave per row) -------
__global__ void k_escore(const __bf16* __restrict__ att1, const float* __restrict__ att2,
                         const float* __restrict__ faw, const float* __restrict__ fab,
                         float* __restrict__ e)
{
    int gw = blockIdx.x * 8 + (threadIdx.x >> 5);
    int lane = threadIdx.x & 31;
    if (gw >= NB * NP) return;
    int b = gw / NP;
    const __bf16* row = att1 + (size_t)gw * NA;
    const float* a2 = att2 + (size_t)b * NA;
    float s = 0.0f;
    for (int a = lane; a < NA; a += 32) {
        float t = (float)row[a] + a2[a];
        s += (t > 0.0f ? t : 0.0f) * faw[a];
    }
#pragma unroll
    for (int off = 16; off; off >>= 1) s += __shfl_xor(s, off, 32);
    if (lane == 0) e[gw] = s + fab[0];
}

// ---- softmax over P, emit masked alphas to d_out ------------------------
__global__ void k_softmax(const float* __restrict__ e, float* __restrict__ alpha,
                          float* __restrict__ out_alphas, const int* __restrict__ declen, int t)
{
    __shared__ float red[256];
    int b = blockIdx.x, tid = threadIdx.x;
    float v = (tid < NP) ? e[b * NP + tid] : -3.0e38f;
    red[tid] = v;
    __syncthreads();
    for (int s = 128; s; s >>= 1) { if (tid < s) red[tid] = fmaxf(red[tid], red[tid + s]); __syncthreads(); }
    float mx = red[0];
    __syncthreads();
    float ex = (tid < NP) ? __expf(v - mx) : 0.0f;
    red[tid] = ex;
    __syncthreads();
    for (int s = 128; s; s >>= 1) { if (tid < s) red[tid] += red[tid + s]; __syncthreads(); }
    float inv = 1.0f / red[0];
    if (tid < NP) {
        float al = ex * inv;
        alpha[b * NP + tid] = al;
        bool act = t < declen[b];
        out_alphas[((size_t)b * NT + t) * NP + tid] = act ? al : 0.0f;
    }
}

// ---- awe = gate * (alpha @ enc); write into xh[:, 512:2560] as bf16 -----
__global__ void k_awe(const float* __restrict__ alpha, const __bf16* __restrict__ enc_b,
                      const float* __restrict__ gbuf, __bf16* __restrict__ xh)
{
    __shared__ float sal[NP];
    int b = blockIdx.x, tid = threadIdx.x;
    if (tid < NP) sal[tid] = alpha[b * NP + tid];
    __syncthreads();
    int e = blockIdx.y * 256 + tid;
    const __bf16* col = enc_b + (size_t)b * NP * NE + e;
    float s = 0.0f;
    for (int p = 0; p < NP; ++p) s += sal[p] * (float)col[(size_t)p * NE];
    float g = gbuf[(size_t)b * NE + e];
    xh[(size_t)b * KXH + NED + e] = (__bf16)(g * s);
}

// ---- fill xh[:, :512]=emb(token) and xh[:, 2560:]=h ---------------------
__global__ void k_xh(const float* __restrict__ emb_w, const int* __restrict__ caps_i,
                     const __bf16* __restrict__ hb, __bf16* __restrict__ xh, int t)
{
    int i = blockIdx.x * 256 + threadIdx.x;   // NB*1024
    int b = i >> 10, j = i & 1023;
    if (j < NED) {
        int tok = caps_i[b * NL + t];
        xh[(size_t)b * KXH + j] = (__bf16)emb_w[(size_t)tok * NED + j];
    } else {
        int n = j - NED;
        xh[(size_t)b * KXH + NED + NE + n] = hb[(size_t)b * ND + n];
    }
}

// ---- LSTM pointwise + ragged masking ------------------------------------
__global__ void k_lstm(const float* __restrict__ gates, float* __restrict__ c,
                       __bf16* __restrict__ hb, __bf16* __restrict__ hnb,
                       const int* __restrict__ declen, int t)
{
    int i = blockIdx.x * 256 + threadIdx.x;   // NB*ND
    int b = i / ND, n = i % ND;
    const float* g = gates + (size_t)b * G4;
    float gi = g[n], gf = g[n + ND], gg = g[n + 2 * ND], go = g[n + 3 * ND];
    float co = c[i], ho = (float)hb[i];
    float cn = sigf(gf) * co + sigf(gi) * tanhf(gg);
    float hn = sigf(go) * tanhf(cn);
    bool act = t < declen[b];
    c[i] = act ? cn : co;
    hb[i] = (__bf16)(act ? hn : ho);
    hnb[i] = (__bf16)hn;
}

// -------------------------------------------------------------------------
extern "C" void kernel_launch(void* const* d_in, const int* in_sizes, int n_in,
                              void* d_out, int out_size, void* d_ws, size_t ws_size,
                              hipStream_t stream)
{
    (void)in_sizes; (void)n_in; (void)out_size; (void)ws_size;
    const float* enc   = (const float*)d_in[0];
    const int*   caps  = (const int*)d_in[1];
    const int*   clen  = (const int*)d_in[2];
    const float* emb_w = (const float*)d_in[3];
    const float* Wih   = (const float*)d_in[4];
    const float* Whh   = (const float*)d_in[5];
    const float* bih   = (const float*)d_in[6];
    const float* bhh   = (const float*)d_in[7];
    const float* h0w   = (const float*)d_in[8];
    const float* h0b   = (const float*)d_in[9];
    const float* c0w   = (const float*)d_in[10];
    const float* c0b   = (const float*)d_in[11];
    const float* gw    = (const float*)d_in[12];
    const float* gb    = (const float*)d_in[13];
    const float* lw    = (const float*)d_in[14];
    const float* lb    = (const float*)d_in[15];
    const float* eaw   = (const float*)d_in[16];
    const float* eab   = (const float*)d_in[17];
    const float* daw   = (const float*)d_in[18];
    const float* dab   = (const float*)d_in[19];
    const float* faw   = (const float*)d_in[20];
    const float* fab   = (const float*)d_in[21];

    // workspace carve (256B aligned)
    char* p = (char*)d_ws;
    auto alloc = [&](size_t bytes) -> char* {
        char* r = p; p += (bytes + 255) & ~(size_t)255; return r;
    };
    __bf16* enc_b  = (__bf16*)alloc((size_t)NB * NP * NE * 2); // 102.8 MB (L2-resident)
    __bf16* att1b  = (__bf16*)alloc((size_t)NB * NP * NA * 2); // 25.7 MB
    __bf16* Wpk    = (__bf16*)alloc((size_t)G4 * KXH * 2);
    __bf16* linb   = (__bf16*)alloc((size_t)NV * ND * 2);
    __bf16* eawb   = (__bf16*)alloc((size_t)NA * NE * 2);
    __bf16* gwb    = (__bf16*)alloc((size_t)NE * ND * 2);
    __bf16* dawb   = (__bf16*)alloc((size_t)NA * ND * 2);
    __bf16* h0wb   = (__bf16*)alloc((size_t)ND * NE * 2);
    __bf16* c0wb   = (__bf16*)alloc((size_t)ND * NE * 2);
    __bf16* meanb  = (__bf16*)alloc((size_t)NB * NE * 2);
    __bf16* hbuf   = (__bf16*)alloc((size_t)NB * ND * 2);
    __bf16* hnew   = (__bf16*)alloc((size_t)NB * ND * 2);
    __bf16* xh     = (__bf16*)alloc((size_t)NB * KXH * 2);
    float*  cst    = (float*)alloc((size_t)NB * ND * 4);
    float*  att2   = (float*)alloc((size_t)NB * NA * 4);
    float*  gbuf   = (float*)alloc((size_t)NB * NE * 4);
    float*  gates  = (float*)alloc((size_t)NB * G4 * 4);
    float*  ebuf   = (float*)alloc((size_t)NB * NP * 4);
    float*  alpha  = (float*)alloc((size_t)NB * NP * 4);
    float*  bsum   = (float*)alloc((size_t)G4 * 4);
    int*    ord_i  = (int*)alloc(NB * 4);
    int*    dl_i   = (int*)alloc(NB * 4);
    int*    caps_i = (int*)alloc((size_t)NB * NL * 4);

    // output layout: pred | caps | dec_len | alphas | order (flat, float)
    float* out       = (float*)d_out;
    float* out_pred  = out;
    float* out_caps  = out_pred + (size_t)NB * NT * NV;
    float* out_dl    = out_caps + (size_t)NB * NL;
    float* out_alph  = out_dl + NB;
    float* out_order = out_alph + (size_t)NB * NT * NP;

    // ---- setup ----
    k_sort<<<1, NB, 0, stream>>>(clen, caps, ord_i, dl_i, caps_i, out_caps, out_dl, out_order);
    k_cvt<<<(NV * ND + 255) / 256, 256, 0, stream>>>(lw, linb, NV * ND);
    k_cvt<<<(NA * NE + 255) / 256, 256, 0, stream>>>(eaw, eawb, NA * NE);
    k_cvt<<<(NE * ND + 255) / 256, 256, 0, stream>>>(gw, gwb, NE * ND);
    k_cvt<<<(NA * ND + 255) / 256, 256, 0, stream>>>(daw, dawb, NA * ND);
    k_cvt<<<(ND * NE + 255) / 256, 256, 0, stream>>>(h0w, h0wb, ND * NE);
    k_cvt<<<(ND * NE + 255) / 256, 256, 0, stream>>>(c0w, c0wb, ND * NE);
    k_pack_w<<<(G4 * KXH) / 256, 256, 0, stream>>>(Wih, Whh, Wpk);
    k_bsum<<<(G4 + 255) / 256, 256, 0, stream>>>(bih, bhh, bsum, G4);
    k_enc<<<(int)(((size_t)NB * NP * NE) / 256), 256, 0, stream>>>(enc, ord_i, enc_b);
    k_mean<<<dim3(NB, NE / 256), 256, 0, stream>>>(enc, ord_i, meanb);

    // h0 (bf16 state), c0 (f32 state)  : N=512 -> grid.y = 2
    k_gemm<<<dim3(NB / 16, 2), 256, 0, stream>>>(meanb, h0wb, h0b, nullptr, hbuf,
        NB, ND, NE, ND, 0, EPI_BF16, nullptr, 0);
    k_gemm<<<dim3(NB / 16, 2), 256, 0, stream>>>(meanb, c0wb, c0b, cst, nullptr,
        NB, ND, NE, ND, 0, EPI_F32, nullptr, 0);

    // att1 = enc_sorted @ ea_w^T  (single 25088x512x2048 bf16 WMMA GEMM)
    k_gemm<<<dim3((NB * NP) / 16, 2), 256, 0, stream>>>(enc_b, eawb, eab, nullptr, att1b,
        NB * NP, NA, NE, 0, 0, EPI_BF16, nullptr, 0);

    // ---- decode loop ----
    for (int t = 0; t < NT; ++t) {
        // att2 = h @ da_w^T + da_b
        k_gemm<<<dim3(NB / 16, 2), 256, 0, stream>>>(hbuf, dawb, dab, att2, nullptr,
            NB, NA, ND, NA, 0, EPI_F32, nullptr, 0);
        // g = sigmoid(h @ gate_w^T + gate_b)   (fused epilogue)
        k_gemm<<<dim3(NB / 16, NE / 256), 256, 0, stream>>>(hbuf, gwb, gb, gbuf, nullptr,
            NB, NE, ND, NE, 0, EPI_SIG, nullptr, 0);
        // attention scores + softmax + context
        k_escore<<<(NB * NP) / 8, 256, 0, stream>>>(att1b, att2, faw, fab, ebuf);
        k_softmax<<<NB, 256, 0, stream>>>(ebuf, alpha, out_alph, dl_i, t);
        k_awe<<<dim3(NB, NE / 256), 256, 0, stream>>>(alpha, enc_b, gbuf, xh);
        k_xh<<<(NB * 1024) / 256, 256, 0, stream>>>(emb_w, caps_i, hbuf, xh, t);
        // gates = [emb, g*awe, h] @ [Wih|Whh]^T + (bih+bhh)
        k_gemm<<<dim3(NB / 16, G4 / 256), 256, 0, stream>>>(xh, Wpk, bsum, gates, nullptr,
            NB, G4, KXH, G4, 0, EPI_F32, nullptr, 0);
        k_lstm<<<(NB * ND) / 256, 256, 0, stream>>>(gates, cst, hbuf, hnew, dl_i, t);
        // pred = mask(h_new @ lin_w^T + lin_b) strided into d_out[b, t, :]
        // N=10000 -> 625 tiles -> 40 y-blocks (per-tile guards)
        k_gemm<<<dim3(NB / 16, (NV / 16 + 15) / 16), 256, 0, stream>>>(hnew, linb, lb,
            out_pred, nullptr, NB, NV, ND, (long)NT * NV, (long)t * NV, EPI_F32, dl_i, t);
    }
}
